// RotationConditionerBlock_22222160789919
// MI455X (gfx1250) — matrix operand order repaired and checked by reading, not compiled, not run on hardware
//
#include <hip/hip_runtime.h>
#include <hip/hip_bf16.h>

// ---------------------------------------------------------------------------
// Problem sizes (fixed by the reference)
// ---------------------------------------------------------------------------
constexpr int SEQ    = 2048;   // S
constexpr int DMODEL = 256;    // D
constexpr int CDIM   = 64;     // channels per head
constexpr float PI2  = 6.28318530717958647692f;

typedef __attribute__((ext_vector_type(16))) __bf16 v16bf;
typedef __attribute__((ext_vector_type(8)))  __bf16 v8bf;
typedef __attribute__((ext_vector_type(8)))  float  v8f;
typedef __attribute__((ext_vector_type(4)))  unsigned int v4u;
typedef __attribute__((ext_vector_type(8)))  int  v8i;
typedef __attribute__((ext_vector_type(4)))  int  v4i;

// ---------------------------------------------------------------------------
// WMMA helper (v_wmma_f32_16x16x32_bf16)
// ---------------------------------------------------------------------------
__device__ __forceinline__ v8f wmma_bf16(v16bf a, v16bf b, v8f c) {
  return __builtin_amdgcn_wmma_f32_16x16x32_bf16(false, a, false, b, (short)0, c,
                                                 false, false);
}

// A-fragment 16x32 bf16 (M = lane&15) from row-major [row][stride].
// Element e of lane: K = col0 + hi*16 + e
__device__ __forceinline__ v16bf load_a_frag(const __bf16* base, int stride,
                                             int row, int col0, int hi) {
  const __bf16* p0 = base + (size_t)row * stride + col0 + hi * 16;
  v8bf x0 = *(const v8bf*)p0;
  v8bf x1 = *(const v8bf*)(p0 + 8);
  v16bf r;
#pragma unroll
  for (int i = 0; i < 8; ++i) { r[i] = x0[i]; r[i + 8] = x1[i]; }
  return r;
}

// B-fragment 32x16 bf16: element e of lane: K = hi*16 + e, N = lane&15.
__device__ __forceinline__ v16bf load_b_contig(const __bf16* p) {
  v8bf x0 = *(const v8bf*)p;
  v8bf x1 = *(const v8bf*)(p + 8);
  v16bf r;
#pragma unroll
  for (int i = 0; i < 8; ++i) { r[i] = x0[i]; r[i + 8] = x1[i]; }
  return r;
}

// B-fragment where K advances by `stride` elements (row-major V tile in LDS).
__device__ __forceinline__ v16bf load_b_strided(const __bf16* p, int stride) {
  v16bf r;
#pragma unroll
  for (int e = 0; e < 16; ++e) r[e] = p[(size_t)e * stride];
  return r;
}

// ---------------------------------------------------------------------------
// Tensor Data Mover: DMA one 32x256 bf16 tile (rows of V) into LDS.
// D# per CDNA5 ISA; 2-D tensor -> groups 2/3 = 0.
// ---------------------------------------------------------------------------
__device__ __forceinline__ void tdm_load_v_tile(const __bf16* gsrc,
                                                __bf16* lds_dst) {
  unsigned long long ga = (unsigned long long)(uintptr_t)gsrc;
  unsigned int lds_off  = (unsigned int)(uintptr_t)lds_dst;  // addr[31:0] = LDS offset
  v4u g0;
  g0[0] = 1u;                                            // count=1, user desc
  g0[1] = lds_off;                                       // lds_addr
  g0[2] = (unsigned int)(ga & 0xffffffffu);              // global_addr[31:0]
  g0[3] = (unsigned int)((ga >> 32) & 0x1ffffffu)        // global_addr[56:32]
          | (2u << 30);                                  // type = 2 ("image")
  v8i g1;
  g1[0] = (int)(1u << 16);     // workgroup_mask=0 | data_size=1 (2 bytes)
  g1[1] = (int)(256u << 16);   // atomic_barrier_addr=0 | tensor_dim0[15:0]=256
  g1[2] = (int)(2048u << 16);  // tensor_dim0[31:16]=0 | tensor_dim1[15:0]=2048
  g1[3] = (int)(256u << 16);   // tensor_dim1[31:16]=0 | tile_dim0=256
  g1[4] = 32;                  // tile_dim1=32 | tile_dim2=0
  g1[5] = 256;                 // tensor_dim0_stride[31:0]=256
  g1[6] = 0;                   // dim0_stride[47:32]=0 | dim1_stride[15:0]=0
  g1[7] = 0;
  v4i z4 = {0, 0, 0, 0};
#if defined(__clang_major__) && __clang_major__ >= 23
  v8i z8 = {0, 0, 0, 0, 0, 0, 0, 0};
  __builtin_amdgcn_tensor_load_to_lds(g0, g1, z4, z4, z8, 0);
#else
  __builtin_amdgcn_tensor_load_to_lds(g0, g1, z4, z4, 0);
#endif
}

// ---------------------------------------------------------------------------
// Kernel 1: f32 -> bf16 conversion
// ---------------------------------------------------------------------------
__global__ void cvt_bf16(const float* __restrict__ src, __bf16* __restrict__ dst,
                         int n) {
  int i = blockIdx.x * blockDim.x + threadIdx.x;
  if (i < n) dst[i] = (__bf16)src[i];
}

// ---------------------------------------------------------------------------
// Kernel 2: tiny projections (pos: inner 6, aux: inner 16) -> heads 8..23
// ---------------------------------------------------------------------------
__global__ void proj_small(const float* __restrict__ pos,
                           const float* __restrict__ aux,
                           const float* __restrict__ Wp,  // (1024,6)
                           const float* __restrict__ bp,  // (1024,)
                           const float* __restrict__ Wa,  // (1024,16)
                           __bf16* __restrict__ Kbf, __bf16* __restrict__ Qbf) {
  int idx = blockIdx.x * blockDim.x + threadIdx.x;  // SEQ*16*64
  int s   = idx >> 10;
  int rem = idx & 1023;
  int hp  = rem >> 6;   // 0..15
  int c   = rem & 63;
  float kv, qv;
  int h = 8 + hp;       // pos heads 8..15, aux heads 16..23
  if (hp < 8) {
    float pf[6];
#pragma unroll
    for (int t = 0; t < 3; ++t) {
      float p = PI2 * pos[s * 3 + t];
      pf[t]     = cosf(p);
      pf[t + 3] = sinf(p);
    }
    int rk = hp * 128 + c, rq = rk + 64;
    kv = bp[rk]; qv = bp[rq];
#pragma unroll
    for (int t = 0; t < 6; ++t) {
      kv += pf[t] * Wp[rk * 6 + t];
      qv += pf[t] * Wp[rq * 6 + t];
    }
  } else {
    int ha = hp - 8;
    int rk = ha * 128 + c, rq = rk + 64;
    kv = 0.f; qv = 0.f;
#pragma unroll
    for (int t = 0; t < 16; ++t) {
      float a = aux[s * 16 + t];
      kv += a * Wa[rk * 16 + t];
      qv += a * Wa[rq * 16 + t];
    }
  }
  Kbf[((size_t)h * SEQ + s) * CDIM + c] = (__bf16)kv;
  Qbf[((size_t)h * SEQ + s) * CDIM + c] = (__bf16)qv;
}

// ---------------------------------------------------------------------------
// Kernel 3: WMMA GEMM  out = A(2048x256) @ W^T + bias
// mode 0: W_nodes_kq (1024x256) -> K/Q heads 0..7;  mode 1: W_val -> V
// ---------------------------------------------------------------------------
__global__ void gemm_proj(const __bf16* __restrict__ A,
                          const __bf16* __restrict__ W,
                          const float* __restrict__ bias,
                          __bf16* __restrict__ Kbf, __bf16* __restrict__ Qbf,
                          __bf16* __restrict__ Vbf, int mode) {
  int tid  = threadIdx.x;
  int wave = tid >> 5;
  int lane = tid & 31;
  int hi   = lane >> 4;
  int ln   = lane & 15;
  int s0   = blockIdx.x * 16;
  int n0   = (blockIdx.y * 4 + wave) * 16;
  int r    = n0 + ln;

  float bvv = bias[r];
  v8f c;
#pragma unroll
  for (int v = 0; v < 8; ++v) c[v] = bvv;

#pragma unroll
  for (int kc = 0; kc < DMODEL; kc += 32) {
    v16bf a = load_a_frag(A, DMODEL, s0 + ln, kc, hi);
    v16bf b = load_b_contig(W + (size_t)r * DMODEL + kc + hi * 16);
    c = wmma_bf16(a, b, c);
  }

#pragma unroll
  for (int v = 0; v < 8; ++v) {
    int s = s0 + v + 8 * hi;
    __bf16 val = (__bf16)c[v];
    if (mode == 0) {
      int hh = r >> 7, rem = r & 127;
      if (rem < 64)
        Kbf[((size_t)hh * SEQ + s) * CDIM + rem] = val;
      else
        Qbf[((size_t)hh * SEQ + s) * CDIM + (rem - 64)] = val;
    } else {
      int g = r >> 8, d = r & 255;
      Vbf[((size_t)g * SEQ + s) * DMODEL + d] = val;
    }
  }
}

// ---------------------------------------------------------------------------
// Kernel 4: flash attention.  Grid (SEQ/16, 8); block = 1 wave.
// V j-tiles are DMA'd into LDS by the Tensor Data Mover (double buffered,
// synchronized with s_wait_tensorcnt) and consumed as B-fragments from LDS.
// ---------------------------------------------------------------------------
__global__ void attn(const __bf16* __restrict__ Kbf,
                     const __bf16* __restrict__ Qbf,
                     const __bf16* __restrict__ Vbf,
                     float* __restrict__ part) {
  __shared__ __bf16 vtile[2][32 * DMODEL];  // 2 x 16 KB V staging
  __shared__ __bf16 pbuf[16 * 32];          // P tile scratch
  int i0   = blockIdx.x * 16;
  int hg   = blockIdx.y;                    // head group 0..7
  int lane = threadIdx.x & 31;
  int hi   = lane >> 4;
  int ln   = lane & 15;

  v8f acc[16];
#pragma unroll
  for (int t = 0; t < 16; ++t)
#pragma unroll
    for (int v = 0; v < 8; ++v) acc[t][v] = 0.f;

#pragma unroll 1
  for (int hh = 0; hh < 3; ++hh) {
    int h = hg * 3 + hh;
    const __bf16* Kh = Kbf + (size_t)h * SEQ * CDIM;
    const __bf16* Qh = Qbf + (size_t)h * SEQ * CDIM;
    const __bf16* Vh = Vbf + (size_t)h * SEQ * DMODEL;

    // kick off the first V tile DMA, then load K while it flies
    tdm_load_v_tile(Vh, &vtile[0][0]);

    v16bf ka0 = load_a_frag(Kh, CDIM, i0 + ln, 0, hi);
    v16bf ka1 = load_a_frag(Kh, CDIM, i0 + ln, 32, hi);

    float m[8], l[8];
#pragma unroll
    for (int v = 0; v < 8; ++v) { m[v] = -__builtin_inff(); l[v] = 0.f; }
    v8f o[16];
#pragma unroll
    for (int t = 0; t < 16; ++t)
#pragma unroll
      for (int v = 0; v < 8; ++v) o[t][v] = 0.f;

#pragma unroll 1
    for (int j0 = 0; j0 < SEQ; j0 += 32) {
      int cur = (j0 >> 5) & 1;
      // prefetch + async-DMA the next V tile into the other buffer
      if (j0 + 32 < SEQ) {
        __builtin_prefetch(Qh + (size_t)(j0 + 32) * CDIM, 0, 3);
        asm volatile("s_wait_dscnt 0" ::: "memory");  // buffer reuse guard
        tdm_load_v_tile(Vh + (size_t)(j0 + 32) * DMODEL, &vtile[cur ^ 1][0]);
      }

      // ---- logits L[i, j0..j0+31] : two 16x16 tiles, K-dim = 64 channels
      v8f L0 = {0.f,0.f,0.f,0.f,0.f,0.f,0.f,0.f};
      v8f L1 = {0.f,0.f,0.f,0.f,0.f,0.f,0.f,0.f};
      {
        v16bf qb = load_b_contig(Qh + (size_t)(j0 + ln) * CDIM + hi * 16);
        L0 = wmma_bf16(ka0, qb, L0);
        qb = load_b_contig(Qh + (size_t)(j0 + ln) * CDIM + 32 + hi * 16);
        L0 = wmma_bf16(ka1, qb, L0);
        qb = load_b_contig(Qh + (size_t)(j0 + 16 + ln) * CDIM + hi * 16);
        L1 = wmma_bf16(ka0, qb, L1);
        qb = load_b_contig(Qh + (size_t)(j0 + 16 + ln) * CDIM + 32 + hi * 16);
        L1 = wmma_bf16(ka1, qb, L1);
      }
      // ---- online softmax over j (rows replicated across 16-lane groups)
      float tmax[8];
#pragma unroll
      for (int v = 0; v < 8; ++v) tmax[v] = fmaxf(L0[v], L1[v]);
#pragma unroll
      for (int mask = 1; mask < 16; mask <<= 1)
#pragma unroll
        for (int v = 0; v < 8; ++v)
          tmax[v] = fmaxf(tmax[v], __shfl_xor(tmax[v], mask, 32));

      float alpha[8];
#pragma unroll
      for (int v = 0; v < 8; ++v) {
        float mn = fmaxf(m[v], tmax[v]);
        alpha[v] = __expf(m[v] - mn);
        m[v] = mn;
      }
#pragma unroll
      for (int t = 0; t < 16; ++t)
#pragma unroll
        for (int v = 0; v < 8; ++v) o[t][v] *= alpha[v];

      float p0[8], p1[8], rs[8];
#pragma unroll
      for (int v = 0; v < 8; ++v) {
        p0[v] = __expf(L0[v] - m[v]);
        p1[v] = __expf(L1[v] - m[v]);
        rs[v] = p0[v] + p1[v];
      }
#pragma unroll
      for (int mask = 1; mask < 16; mask <<= 1)
#pragma unroll
        for (int v = 0; v < 8; ++v) rs[v] += __shfl_xor(rs[v], mask, 32);
#pragma unroll
      for (int v = 0; v < 8; ++v) l[v] = l[v] * alpha[v] + rs[v];

      // ---- P (16x32) through LDS: C-layout stores -> A-layout loads
#pragma unroll
      for (int v = 0; v < 8; ++v) {
        int row = v + 8 * hi;
        pbuf[row * 32 + ln]      = (__bf16)p0[v];
        pbuf[row * 32 + 16 + ln] = (__bf16)p1[v];
      }
      asm volatile("s_wait_dscnt 0" ::: "memory");
      v16bf pa = load_a_frag(pbuf, 32, ln, 0, hi);

      // ---- wait for this j-tile's V DMA (<=1 outstanding keeps next in flight)
      if (j0 + 32 < SEQ)
        __builtin_amdgcn_s_wait_tensorcnt(1);
      else
        __builtin_amdgcn_s_wait_tensorcnt(0);

      // ---- O += P @ V from the LDS-resident tile (16 d-tiles)
      const __bf16* vbase = &vtile[cur][0] + (size_t)(hi * 16) * DMODEL + ln;
#pragma unroll
      for (int dt = 0; dt < 16; ++dt) {
        v16bf vb = load_b_strided(vbase + dt * 16, DMODEL);
        o[dt] = wmma_bf16(pa, vb, o[dt]);
      }
    }

    // ---- normalize and accumulate over this wave's heads
    float inv[8];
#pragma unroll
    for (int v = 0; v < 8; ++v) inv[v] = 1.f / l[v];
#pragma unroll
    for (int t = 0; t < 16; ++t)
#pragma unroll
      for (int v = 0; v < 8; ++v) acc[t][v] += o[t][v] * inv[v];
  }

  float* pout = part + ((size_t)hg * SEQ + i0) * DMODEL;
#pragma unroll
  for (int t = 0; t < 16; ++t)
#pragma unroll
    for (int v = 0; v < 8; ++v)
      pout[(v + 8 * hi) * DMODEL + t * 16 + ln] = acc[t][v];
}

// ---------------------------------------------------------------------------
// Kernel 5: deterministic reduction over the 8 head-group partials
// ---------------------------------------------------------------------------
__global__ void reduce8(const float* __restrict__ part, float* __restrict__ out) {
  int i = blockIdx.x;
  int d = threadIdx.x;
  float s = 0.f;
#pragma unroll
  for (int w = 0; w < 8; ++w) s += part[((size_t)w * SEQ + i) * DMODEL + d];
  out[(size_t)i * DMODEL + d] = s;
}

// ---------------------------------------------------------------------------
// Launch
// ---------------------------------------------------------------------------
extern "C" void kernel_launch(void* const* d_in, const int* in_sizes, int n_in,
                              void* d_out, int out_size, void* d_ws,
                              size_t ws_size, hipStream_t stream) {
  (void)in_sizes; (void)n_in; (void)out_size; (void)ws_size;
  const float* nodes = (const float*)d_in[0];
  const float* pos   = (const float*)d_in[1];
  const float* aux   = (const float*)d_in[2];
  const float* Wn    = (const float*)d_in[3];  // (1024,256)
  const float* bn    = (const float*)d_in[4];
  const float* Wp    = (const float*)d_in[5];  // (1024,6)
  const float* bp    = (const float*)d_in[6];
  const float* Wa    = (const float*)d_in[7];  // (1024,16)
  const float* Wv    = (const float*)d_in[8];  // (6144,256)
  const float* bv    = (const float*)d_in[9];
  float* out = (float*)d_out;

  // workspace layout (bytes)
  char* w = (char*)d_ws;
  __bf16* nodes_bf = (__bf16*)(w + 0);          // 1,048,576
  __bf16* Wkq_bf   = (__bf16*)(w + 1048576);    //   524,288
  __bf16* Wv_bf    = (__bf16*)(w + 1572864);    // 3,145,728
  __bf16* Kbf      = (__bf16*)(w + 4718592);    // 6,291,456
  __bf16* Qbf      = (__bf16*)(w + 11010048);   // 6,291,456
  __bf16* Vbf      = (__bf16*)(w + 17301504);   // 25,165,824
  float*  part     = (float*)(w + 42467328);    // 16,777,216

  cvt_bf16<<<(SEQ * DMODEL + 255) / 256, 256, 0, stream>>>(nodes, nodes_bf,
                                                           SEQ * DMODEL);
  cvt_bf16<<<(1024 * 256 + 255) / 256, 256, 0, stream>>>(Wn, Wkq_bf, 1024 * 256);
  cvt_bf16<<<(6144 * 256 + 255) / 256, 256, 0, stream>>>(Wv, Wv_bf, 6144 * 256);

  proj_small<<<(SEQ * 16 * 64) / 256, 256, 0, stream>>>(pos, aux, Wp, bp, Wa,
                                                        Kbf, Qbf);

  gemm_proj<<<dim3(SEQ / 16, 1024 / 64), 128, 0, stream>>>(nodes_bf, Wkq_bf, bn,
                                                           Kbf, Qbf, Vbf, 0);
  gemm_proj<<<dim3(SEQ / 16, 6144 / 64), 128, 0, stream>>>(nodes_bf, Wv_bf, bv,
                                                           Kbf, Qbf, Vbf, 1);

  attn<<<dim3(SEQ / 16, 8), 32, 0, stream>>>(Kbf, Qbf, Vbf, part);

  reduce8<<<SEQ, DMODEL, 0, stream>>>(part, out);
}
